// TagSequenceModel_46084999086673
// MI455X (gfx1250) — compile-verified, matrix operand order checked by reading, not run on hardware
//
#include <hip/hip_runtime.h>
#include <hip/hip_bf16.h>

// Problem constants (from the reference)
#define B_ 128
#define T_ 512
#define N_ 200
#define START_I 1
#define END_I 2
#define NEG_INF -10000.0f

// Tiling: N padded to 13*16 = 208 columns, K padded to 7*32 = 224
#define NT 13
#define KT 7
#define KPAD 224

typedef __attribute__((ext_vector_type(16))) __bf16 v16bf;
typedef __attribute__((ext_vector_type(8)))  float  v8f;

union Frag {
    v16bf v;
    uint4 q[2];
    unsigned short s[16];
};

static __device__ __forceinline__ unsigned short f2bf(float x) {
    // round-to-nearest-even float -> bf16 bits
    unsigned int u = __float_as_uint(x);
    u = (u + 0x7FFFu + ((u >> 16) & 1u)) >> 16;
    return (unsigned short)u;
}

// ---------------------------------------------------------------------------
// Kernel 1: build per-lane WMMA B-fragment images of E^T, E = exp(trans).
// B[k][n] = exp(trans[n*N + k]).  One block (32 lanes) per (jt,kt) fragment.
// Layout per CDNA5 ISA 16-bit matrix table (lane = N column; VGPR pairs
// hold K {h*8+0..7, 16+h*8+0..7} where h = lane/16).
// ---------------------------------------------------------------------------
__global__ void crf_prep_bfrags(const float* __restrict__ trans,
                                unsigned short* __restrict__ bfr) {
    const int blk  = blockIdx.x;        // 0 .. NT*KT-1
    const int jt   = blk / KT;
    const int kt   = blk % KT;
    const int lane = threadIdx.x;       // 0..31
    const int h    = lane >> 4;
    const int n    = jt * 16 + (lane & 15);

    __align__(16) unsigned short out[16];
#pragma unroll
    for (int s = 0; s < 16; ++s) {
        const int v = s >> 1, p = s & 1;
        const int kl = (v < 4) ? (h * 8 + 2 * v + p)
                               : (16 + h * 8 + 2 * (v - 4) + p);
        const int k = kt * 32 + kl;
        float val = 0.0f;                              // zero padding: no contribution
        if (n < N_ && k < N_) val = __expf(trans[n * N_ + k]);
        out[s] = f2bf(val);
    }
    unsigned short* dst = bfr + (size_t)(blk * 32 + lane) * 16;
    ((uint4*)dst)[0] = *(const uint4*)&out[0];
    ((uint4*)dst)[1] = *(const uint4*)&out[8];
}

// ---------------------------------------------------------------------------
// Kernel 2: CRF forward recursion.  grid = B/16 workgroups, 13 waves each.
// Wave jt owns output N-tile jt; its 7 B-fragments stay in VGPRs for the
// entire T loop.  Per step (2 barriers):
//   fold: every lane folds 13 tile maxima -> running shift m (register)
//   (a)   P = exp(alpha - m) as bf16 in LDS (row-major)
//   (b)   preload all 7 A fragments + unconditional unary loads
//   (c)   7x v_wmma_f32_16x16x32_bf16 (f32 accum)
//   (d)   alpha' = u + m + log(C), per-row length mask, tile row-max
// ---------------------------------------------------------------------------
__global__ void __launch_bounds__(NT * 32)
crf_forward(const float* __restrict__ unary,   // [B,T,N]
            const int*   __restrict__ lengths, // [B]
            const float* __restrict__ trans,   // [N,N]
            const unsigned short* __restrict__ bfr,
            float* __restrict__ fwd) {         // [B]
    __shared__ float alphaS[16][KPAD + 1];                 // +1: bank-spread scalars
    __shared__ __align__(16) unsigned short Ps[16][KPAD];  // bf16 P, row-major
    __shared__ float wmaxS[NT][16];                        // per-wave tile row maxima
    __shared__ float rowmaxS[16];                          // LDS mirror of m

    const int tid  = threadIdx.x;
    const int lane = tid & 31;
    const int jt   = tid >> 5;          // wave id == N-tile id
    const int h    = lane >> 4;
    const int r16  = lane & 15;
    const int b0   = blockIdx.x * 16;

    // ---- init alpha0 / P padding ----
    for (int i = tid; i < 16 * KPAD; i += blockDim.x) {
        const int row = i / KPAD, col = i % KPAD;
        alphaS[row][col] = (col == START_I) ? 0.0f : NEG_INF;
        Ps[row][col] = 0;               // cols >= 208 stay 0 forever
    }

    // ---- per-lane register-cached state ----
    const int lenM = lengths[b0 + r16];        // length of "my" m-row
    int lenR[8];
#pragma unroll
    for (int v = 0; v < 8; ++v) lenR[v] = lengths[b0 + v + h * 8];

    const int col   = jt * 16 + r16;
    const bool colOK = (col < N_);
    const int colC  = colOK ? col : 0;         // clamped for unconditional loads
    const float* baseU[8];
#pragma unroll
    for (int v = 0; v < 8; ++v)
        baseU[v] = unary + (size_t)(b0 + v + h * 8) * T_ * N_ + colC;

    float m = 0.0f;                            // running shift for row r16

    // ---- load resident B fragments (constant over all T steps) ----
    Frag bf[KT];
#pragma unroll
    for (int kt = 0; kt < KT; ++kt) {
        const unsigned short* src = bfr + (size_t)((jt * KT + kt) * 32 + lane) * 16;
        bf[kt].q[0] = ((const uint4*)src)[0];
        bf[kt].q[1] = ((const uint4*)src)[1];
    }
    __syncthreads();

    for (int t = 0; t < T_; ++t) {
        // ---- fold previous step's tile maxima into running shift m ----
        if (t > 0) {
            float nm = NEG_INF;
#pragma unroll
            for (int w = 0; w < NT; ++w) nm = fmaxf(nm, wmaxS[w][r16]);
            if ((t - 1) < lenM) m = nm;        // row updated last step -> new max
        }
        if (lane < 16) rowmaxS[r16] = m;       // mirror for step (d)

        // ---- (a) wave jt writes P cols [jt*16, +16): lane -> (row r16, 8 cols)
        {
            const int cbase = jt * 16 + h * 8;
            unsigned int w4[4];
#pragma unroll
            for (int c = 0; c < 4; ++c) {
                const float a0 = alphaS[r16][cbase + 2 * c];
                const float a1 = alphaS[r16][cbase + 2 * c + 1];
                w4[c] = (unsigned int)f2bf(__expf(a0 - m)) |
                        ((unsigned int)f2bf(__expf(a1 - m)) << 16);
            }
            uint4 pkq = make_uint4(w4[0], w4[1], w4[2], w4[3]);
            *(uint4*)&Ps[r16][cbase] = pkq;
        }
        __syncthreads();                       // barrier 1: P visible

        // ---- (b) unconditional unary loads (clamped col) + A preload ----
        float uv[8];
#pragma unroll
        for (int v = 0; v < 8; ++v)
            uv[v] = baseU[v][(size_t)t * N_];

        Frag a[KT];
#pragma unroll
        for (int kt = 0; kt < KT; ++kt) {
            const unsigned short* ap = &Ps[r16][kt * 32 + h * 8];
            a[kt].q[0] = *(const uint4*)ap;        // K = kt*32 + h*8 + 0..7
            a[kt].q[1] = *(const uint4*)(ap + 16); // K = kt*32 + h*8 + 16..23
        }

        // ---- (c) C = P(16x224) x E^T-tile(224x16): 7 chained bf16 WMMAs ----
        v8f c = {};
#pragma unroll
        for (int kt = 0; kt < KT; ++kt)
            c = __builtin_amdgcn_wmma_f32_16x16x32_bf16(
                    false, a[kt].v, false, bf[kt].v, (short)0, c, false, false);

        // ---- (d) alpha' = u + m + log(C), per-row mask; tile row max ----
        float mR[8];
#pragma unroll
        for (int v = 0; v < 8; ++v) mR[v] = rowmaxS[v + h * 8];

        float mx[8];
#pragma unroll
        for (int v = 0; v < 8; ++v) {
            const float nv = uv[v] + mR[v] + __logf(fmaxf(c[v], 1e-37f));
            const bool ok = colOK && (t < lenR[v]);
            if (ok) alphaS[v + h * 8][col] = nv;
            mx[v] = ok ? nv : NEG_INF;
        }
        // butterfly max across the 16 lanes of each half (stays within half)
#pragma unroll
        for (int d = 1; d < 16; d <<= 1) {
#pragma unroll
            for (int v = 0; v < 8; ++v)
                mx[v] = fmaxf(mx[v], __shfl_xor(mx[v], d, 32));
        }
        if (r16 == 0) {
#pragma unroll
            for (int v = 0; v < 8; ++v)
                wmaxS[jt][v + h * 8] = mx[v];
        }
        __syncthreads();                       // barrier 2: alpha/wmax visible
    }

    // ---- terminal: fwd[b] = logsumexp_j(alpha[b,j] + trans[END, j]) ----
    if (tid < 16) {                            // wave 0, r16 == tid, m is valid
        float nm = NEG_INF;
#pragma unroll
        for (int w = 0; w < NT; ++w) nm = fmaxf(nm, wmaxS[w][tid]);
        const float mf = ((T_ - 1) < lenM) ? nm : m;
        float s = 0.0f;
        for (int j = 0; j < N_; ++j)
            s += __expf(alphaS[tid][j] + trans[END_I * N_ + j] - mf);
        fwd[b0 + tid] = mf + __logf(s);
    }
}

// ---------------------------------------------------------------------------
// Kernel 3: gold path score + mean(fwd - gold) -> out[0]
// ---------------------------------------------------------------------------
__global__ void crf_finish(const float* __restrict__ unary,
                           const int*   __restrict__ tags,
                           const int*   __restrict__ lengths,
                           const float* __restrict__ trans,
                           const float* __restrict__ fwd,
                           float* __restrict__ out) {
    __shared__ float red[B_];
    const int b = threadIdx.x;          // 0..127
    const int len = lengths[b];
    float s = 0.0f;
    int prev = START_I;
    for (int t = 0; t < T_; ++t) {
        const int curtag = tags[b * T_ + t];
        if (t < len)
            s += trans[curtag * N_ + prev] +
                 unary[((size_t)b * T_ + t) * N_ + curtag];
        prev = curtag;
    }
    const int last = (len > 0) ? tags[b * T_ + len - 1] : START_I;
    s += trans[END_I * N_ + last];
    red[b] = fwd[b] - s;
    __syncthreads();
    for (int st = B_ / 2; st > 0; st >>= 1) {
        if (b < st) red[b] += red[b + st];
        __syncthreads();
    }
    if (b == 0) out[0] = red[0] / (float)B_;
}

// ---------------------------------------------------------------------------
extern "C" void kernel_launch(void* const* d_in, const int* in_sizes, int n_in,
                              void* d_out, int out_size, void* d_ws, size_t ws_size,
                              hipStream_t stream) {
    const float* unary   = (const float*)d_in[0];  // [B,T,N] f32
    const int*   tags    = (const int*)  d_in[1];  // [B,T] i32
    const int*   lengths = (const int*)  d_in[2];  // [B] i32
    const float* trans   = (const float*)d_in[3];  // [N,N] f32

    // workspace: [ B-fragments: NT*KT*32*16 ushort = 93184 B ][ fwd: 128 f32 ]
    unsigned short* bfr = (unsigned short*)d_ws;
    float* fwd = (float*)((char*)d_ws + (size_t)NT * KT * 32 * 16 * sizeof(unsigned short));

    crf_prep_bfrags<<<NT * KT, 32, 0, stream>>>(trans, bfr);
    crf_forward<<<B_ / 16, NT * 32, 0, stream>>>(unary, lengths, trans, bfr, fwd);
    crf_finish<<<1, B_, 0, stream>>>(unary, tags, lengths, trans, fwd, (float*)d_out);
}
